// PointNetPlusPlusSegmentation_42880953483443
// MI455X (gfx1250) — matrix operand order repaired and checked by reading, not compiled, not run on hardware
//
#include <hip/hip_runtime.h>

// ---------------- types ----------------
typedef _Float16 half_t;
typedef half_t v16h __attribute__((ext_vector_type(16)));
typedef half_t h8v  __attribute__((ext_vector_type(8)));
typedef float  v8f  __attribute__((ext_vector_type(8)));

// Problem constants (from setup_inputs)
#define BB 16
#define NN 4096
#define S1 512
#define K1 32
#define S2 128
#define K2 64

// ---------------- WMMA helpers (wave32, 16x16x32 f16 -> f32) ----------------
// A operand: 16(M) x 32(K) f16, row-major source with row stride `stride` (halves).
// ISA layout: lane<16 holds row M=lane, halves[0..7]=K0..7, halves[8..15]=K16..23;
//             lane>=16 holds row M=lane-16, K8..15 and K24..31.
__device__ __forceinline__ v16h load_A(const half_t* A, int stride, int m0, int k0) {
  int lane = threadIdx.x & 31;
  const half_t* p = A + (size_t)(m0 + (lane & 15)) * stride + k0 + ((lane & 16) ? 8 : 0);
  h8v lo = *(const h8v*)(p);
  h8v hi = *(const h8v*)(p + 16);
  return __builtin_shufflevector(lo, hi, 0,1,2,3,4,5,6,7,8,9,10,11,12,13,14,15);
}

// B operand: 32(K) x 16(N) f16 taken from W^T stored row-major (Cout rows x Cinp cols).
// lane<16 -> column n0+lane, K=k0..k0+7,k0+16..k0+23 ; lane>=16 -> column n0+lane-16, K=k0+8..k0+15,k0+24..k0+31.
__device__ __forceinline__ v16h load_B(const half_t* WT, int stride, int n0, int k0) {
  int lane = threadIdx.x & 31;
  const half_t* p = WT + (size_t)(n0 + (lane & 15)) * stride + k0 + ((lane & 16) ? 8 : 0);
  h8v lo = *(const h8v*)(p);
  h8v hi = *(const h8v*)(p + 16);
  return __builtin_shufflevector(lo, hi, 0,1,2,3,4,5,6,7,8,9,10,11,12,13,14,15);
}

__device__ __forceinline__ v8f wmma16(v16h a, v16h b, v8f c) {
  return __builtin_amdgcn_wmma_f32_16x16x32_f16(false, a, false, b, (short)0, c, false, false);
}

// One dense layer: Out[0..MT*16) x coutP = relu(A @ W + bias), Out written as f16.
template <int MT>
__device__ void mlp_layer(const half_t* A, int strideA,
                          const half_t* WT, int cinP,
                          const float* bias, int coutP, int coutReal,
                          half_t* Out, int strideO, bool relu) {
  int lane = threadIdx.x & 31;
  for (int nt = 0; nt < coutP; nt += 16) {
    v8f c[MT];
#pragma unroll
    for (int m = 0; m < MT; m++) c[m] = (v8f){0.f,0.f,0.f,0.f,0.f,0.f,0.f,0.f};
    for (int kt = 0; kt < cinP; kt += 32) {
      v16h bm = load_B(WT, cinP, nt, kt);
#pragma unroll
      for (int m = 0; m < MT; m++) {
        v16h am = load_A(A, strideA, m * 16, kt);
        c[m] = wmma16(am, bm, c[m]);
      }
    }
    int col = nt + (lane & 15);
    float bb = (col < coutReal) ? bias[col] : 0.f;
    int rsel = (lane & 16) ? 8 : 0;
#pragma unroll
    for (int m = 0; m < MT; m++) {
#pragma unroll
      for (int i = 0; i < 8; i++) {
        float v = c[m][i] + bb;
        if (relu) v = fmaxf(v, 0.f);
        Out[(size_t)(m * 16 + rsel + i) * strideO + col] = (half_t)v;
      }
    }
  }
}

// Final SA layer: relu(A@W+b) then max over all MT*16 rows -> out_g[coutP] (f16).
template <int MT>
__device__ void mlp_layer_maxpool(const half_t* A, int strideA,
                                  const half_t* WT, int cinP,
                                  const float* bias, int coutP, half_t* out_g) {
  int lane = threadIdx.x & 31;
  for (int nt = 0; nt < coutP; nt += 16) {
    v8f c[MT];
#pragma unroll
    for (int m = 0; m < MT; m++) c[m] = (v8f){0.f,0.f,0.f,0.f,0.f,0.f,0.f,0.f};
    for (int kt = 0; kt < cinP; kt += 32) {
      v16h bm = load_B(WT, cinP, nt, kt);
#pragma unroll
      for (int m = 0; m < MT; m++) {
        v16h am = load_A(A, strideA, m * 16, kt);
        c[m] = wmma16(am, bm, c[m]);
      }
    }
    int col = nt + (lane & 15);
    float bb = bias[col];
    float mx = 0.f;  // relu lower bound
#pragma unroll
    for (int m = 0; m < MT; m++)
#pragma unroll
      for (int i = 0; i < 8; i++) mx = fmaxf(mx, c[m][i] + bb);
    mx = fmaxf(mx, __shfl_xor(mx, 16));
    if (lane < 16) out_g[col] = (half_t)mx;
  }
}

// ---------------- weight prep: f32 (Cin x Cout) -> f16 transposed padded (Coutp x Cinp) -----
__global__ void prep_wt_kernel(const float* __restrict__ w, half_t* __restrict__ wt,
                               int Cin, int Cout, int Cinp, int Coutp) {
  int i = blockIdx.x * blockDim.x + threadIdx.x;
  int total = Coutp * Cinp;
  if (i >= total) return;
  int o = i / Cinp, k = i % Cinp;
  float v = (o < Cout && k < Cin) ? w[(size_t)k * Cout + o] : 0.f;
  wt[i] = (half_t)v;
}

// ---------------- farthest point sampling (block per batch) ----------------
__global__ void fps_kernel(const float* __restrict__ xyz, int N, int npoint,
                           int* __restrict__ fps_idx, float* __restrict__ new_xyz) {
  int b = blockIdx.x;
  int t = threadIdx.x;
  const float* pts = xyz + (size_t)b * N * 3;
  const int P = N / 256;
  float dist[16];
#pragma unroll
  for (int i = 0; i < 16; i++) dist[i] = 1e10f;
  __shared__ float sval[256];
  __shared__ int sidx[256];
  __shared__ int sfar;
  if (t == 0) sfar = 0;
  __syncthreads();
  for (int it = 0; it < npoint; ++it) {
    int far = sfar;
    if (t == 0) {
      fps_idx[(size_t)b * npoint + it] = far;
      new_xyz[((size_t)b * npoint + it) * 3 + 0] = pts[far * 3 + 0];
      new_xyz[((size_t)b * npoint + it) * 3 + 1] = pts[far * 3 + 1];
      new_xyz[((size_t)b * npoint + it) * 3 + 2] = pts[far * 3 + 2];
    }
    float cx = pts[far * 3 + 0], cy = pts[far * 3 + 1], cz = pts[far * 3 + 2];
    float best = -1.f;
    int bi = 0;
    for (int i = 0; i < P; i++) {
      int p = t + i * 256;
      float dx = pts[p * 3 + 0] - cx;
      float dy = pts[p * 3 + 1] - cy;
      float dz = pts[p * 3 + 2] - cz;
      float d = dx * dx + dy * dy + dz * dz;
      float nd = fminf(dist[i], d);
      dist[i] = nd;
      if (nd > best) { best = nd; bi = p; }
    }
    sval[t] = best;
    sidx[t] = bi;
    __syncthreads();
    for (int s = 128; s > 0; s >>= 1) {
      if (t < s) {
        float v2 = sval[t + s];
        int i2 = sidx[t + s];
        if (v2 > sval[t] || (v2 == sval[t] && i2 < sidx[t])) { sval[t] = v2; sidx[t] = i2; }
      }
      __syncthreads();
    }
    if (t == 0) sfar = sidx[0];
    __syncthreads();
  }
}

// ---------------- ball query: one wave per center; first nsample indices in order ----
__global__ void ball_query_kernel(const float* __restrict__ xyz, int N,
                                  const float* __restrict__ new_xyz, int S,
                                  float r2, int nsample, int* __restrict__ idx) {
  int g = blockIdx.x;  // b*S + s
  int lane = threadIdx.x;
  int b = g / S;
  const float* pts = xyz + (size_t)b * N * 3;
  float cx = new_xyz[g * 3 + 0], cy = new_xyz[g * 3 + 1], cz = new_xyz[g * 3 + 2];
  int* out = idx + (size_t)g * nsample;
  int cnt = 0;
  int first = 0;
  for (int base = 0; base < N && cnt < nsample; base += 32) {
    int p = base + lane;
    float dx = pts[p * 3 + 0] - cx;
    float dy = pts[p * 3 + 1] - cy;
    float dz = pts[p * 3 + 2] - cz;
    bool in = (dx * dx + dy * dy + dz * dz) <= r2;
    unsigned mask = (unsigned)__ballot(in);
    while (mask && cnt < nsample) {
      int l = __builtin_ctz(mask);
      mask &= mask - 1;
      if (lane == 0) out[cnt] = base + l;
      if (cnt == 0) first = base + l;
      cnt++;
    }
  }
  if (lane == 0) {
    if (cnt == 0) { out[0] = 0; first = 0; cnt = 1; }
    for (int k = cnt; k < nsample; k++) out[k] = first;
  }
}

// ---------------- 3-NN + inverse-distance weights ----------------
__global__ void knn3_kernel(const float* __restrict__ tgt, int St,
                            const float* __restrict__ src, int Ss, int total,
                            int* __restrict__ iout, float* __restrict__ wout) {
  int gid = blockIdx.x * blockDim.x + threadIdx.x;
  if (gid >= total) return;
  int b = gid / St;
  const float* s = src + (size_t)b * Ss * 3;
  float tx = tgt[gid * 3 + 0], ty = tgt[gid * 3 + 1], tz = tgt[gid * 3 + 2];
  float d0 = 1e30f, d1 = 1e30f, d2 = 1e30f;
  int i0 = 0, i1 = 0, i2 = 0;
  for (int j = 0; j < Ss; j++) {
    float dx = s[j * 3 + 0] - tx;
    float dy = s[j * 3 + 1] - ty;
    float dz = s[j * 3 + 2] - tz;
    float d = dx * dx + dy * dy + dz * dz;
    if (d < d0)      { d2 = d1; i2 = i1; d1 = d0; i1 = i0; d0 = d; i0 = j; }
    else if (d < d1) { d2 = d1; i2 = i1; d1 = d; i1 = j; }
    else if (d < d2) { d2 = d; i2 = j; }
  }
  float r0 = 1.f / (fmaxf(d0, 0.f) + 1e-8f);
  float r1 = 1.f / (fmaxf(d1, 0.f) + 1e-8f);
  float r2 = 1.f / (fmaxf(d2, 0.f) + 1e-8f);
  float sum = r0 + r1 + r2;
  iout[gid * 3 + 0] = i0; iout[gid * 3 + 1] = i1; iout[gid * 3 + 2] = i2;
  wout[gid * 3 + 0] = r0 / sum; wout[gid * 3 + 1] = r1 / sum; wout[gid * 3 + 2] = r2 / sum;
}

// ---------------- SA1: gather + 3-layer MLP (pad6->32 ->64 ->64 ->128) + maxpool ------
__global__ void sa1_kernel(const float* __restrict__ xyz, const int* __restrict__ idx,
                           const float* __restrict__ newxyz,
                           const half_t* __restrict__ wt1, const half_t* __restrict__ wt2,
                           const half_t* __restrict__ wt3,
                           const float* __restrict__ b1, const float* __restrict__ b2,
                           const float* __restrict__ b3,
                           half_t* __restrict__ f1_out) {
  __shared__ half_t bufA[4][32 * 64];
  __shared__ half_t bufB[4][32 * 64];
  int wave = threadIdx.x >> 5;
  int lane = threadIdx.x & 31;
  int g = blockIdx.x * 4 + wave;  // group in [0, B*S1)
  int b = g / S1;
  half_t* A0 = bufA[wave];
  {
    int i = idx[(size_t)g * K1 + lane];
    const float* p = xyz + ((size_t)b * NN + i) * 3;
    float cx = newxyz[g * 3 + 0], cy = newxyz[g * 3 + 1], cz = newxyz[g * 3 + 2];
    float px = p[0], py = p[1], pz = p[2];
    half_t* row = A0 + lane * 32;
#pragma unroll
    for (int c = 0; c < 32; c++) row[c] = (half_t)0.f;
    row[0] = (half_t)(px - cx); row[1] = (half_t)(py - cy); row[2] = (half_t)(pz - cz);
    row[3] = (half_t)px;        row[4] = (half_t)py;        row[5] = (half_t)pz;
  }
  __syncthreads();
  mlp_layer<2>(A0, 32, wt1, 32, b1, 64, 64, bufB[wave], 64, true);
  __syncthreads();
  mlp_layer<2>(bufB[wave], 64, wt2, 64, b2, 64, 64, bufA[wave], 64, true);
  __syncthreads();
  mlp_layer_maxpool<2>(bufA[wave], 64, wt3, 64, b3, 128, f1_out + (size_t)g * 128);
}

// ---------------- SA2: gather (3 + f1[128], pad->160) -> 128 -> 128 -> 256 + maxpool --
__global__ void sa2_kernel(const float* __restrict__ xyz1, const half_t* __restrict__ f1,
                           const int* __restrict__ idx, const float* __restrict__ newxyz,
                           const half_t* __restrict__ wt1, const half_t* __restrict__ wt2,
                           const half_t* __restrict__ wt3,
                           const float* __restrict__ b1, const float* __restrict__ b2,
                           const float* __restrict__ b3,
                           half_t* __restrict__ f2_out) {
  __shared__ half_t bufA[64 * 160];
  __shared__ half_t bufB[64 * 128];
  int g = blockIdx.x;  // b*S2 + s
  int b = g / S2;
  int lane = threadIdx.x;
  float cx = newxyz[g * 3 + 0], cy = newxyz[g * 3 + 1], cz = newxyz[g * 3 + 2];
  for (int r = lane; r < K2; r += 32) {
    int i = idx[(size_t)g * K2 + r];
    const float* p = xyz1 + ((size_t)b * S1 + i) * 3;
    half_t* row = bufA + r * 160;
    row[0] = (half_t)(p[0] - cx); row[1] = (half_t)(p[1] - cy); row[2] = (half_t)(p[2] - cz);
    const half_t* f = f1 + ((size_t)b * S1 + i) * 128;
    for (int c = 0; c < 128; c++) row[3 + c] = f[c];
    for (int c = 131; c < 160; c++) row[c] = (half_t)0.f;
  }
  __syncthreads();
  mlp_layer<4>(bufA, 160, wt1, 160, b1, 128, 128, bufB, 128, true);
  __syncthreads();
  mlp_layer<4>(bufB, 128, wt2, 128, b2, 128, 128, bufA, 128, true);  // reuse bufA (stride 128)
  __syncthreads();
  mlp_layer_maxpool<4>(bufA, 128, wt3, 128, b3, 256, f2_out + (size_t)g * 256);
}

// ---------------- FP x-row builders ----------------
__global__ void build_fp1_x_kernel(const half_t* __restrict__ f1, const half_t* __restrict__ f2,
                                   const int* __restrict__ ki, const float* __restrict__ kw,
                                   half_t* __restrict__ x) {
  int i = blockIdx.x * blockDim.x + threadIdx.x;
  const int total = BB * S1 * 384;
  if (i >= total) return;
  int row = i / 384, c = i % 384;
  half_t v;
  if (c < 128) {
    v = f1[(size_t)row * 128 + c];
  } else {
    int b = row / S1;
    int c2 = c - 128;
    const int* id = ki + row * 3;
    const float* w = kw + row * 3;
    float acc = 0.f;
#pragma unroll
    for (int j = 0; j < 3; j++)
      acc += w[j] * (float)f2[((size_t)b * S2 + id[j]) * 256 + c2];
    v = (half_t)acc;
  }
  x[(size_t)row * 384 + c] = v;
}

__global__ void build_fp2_x_kernel(const float* __restrict__ xyz, const half_t* __restrict__ up1,
                                   const int* __restrict__ ki, const float* __restrict__ kw,
                                   half_t* __restrict__ x) {
  int i = blockIdx.x * blockDim.x + threadIdx.x;
  const int total = BB * NN * 160;
  if (i >= total) return;
  int row = i / 160, c = i % 160;
  half_t v;
  if (c < 3) {
    v = (half_t)xyz[(size_t)row * 3 + c];
  } else if (c < 131) {
    int b = row / NN;
    int c2 = c - 3;
    const int* id = ki + row * 3;
    const float* w = kw + row * 3;
    float acc = 0.f;
#pragma unroll
    for (int j = 0; j < 3; j++)
      acc += w[j] * (float)up1[((size_t)b * S1 + id[j]) * 128 + c2];
    v = (half_t)acc;
  } else {
    v = (half_t)0.f;
  }
  x[(size_t)row * 160 + c] = v;
}

// ---------------- FP1 MLP chain: 384 -> 256 -> 128 (relu both) ----------------
__global__ void fp1_kernel(const half_t* __restrict__ x,
                           const half_t* __restrict__ wt1, const half_t* __restrict__ wt2,
                           const float* __restrict__ b1, const float* __restrict__ b2,
                           half_t* __restrict__ up1) {
  __shared__ half_t buf[2][32 * 256];
  int wave = threadIdx.x >> 5;
  int row0 = blockIdx.x * 64 + wave * 32;
  mlp_layer<2>(x + (size_t)row0 * 384, 384, wt1, 384, b1, 256, 256, buf[wave], 256, true);
  __syncthreads();
  mlp_layer<2>(buf[wave], 256, wt2, 256, b2, 128, 128, up1 + (size_t)row0 * 128, 128, true);
}

// ---------------- FP2 + FC chain: 160p(131)->128->128->128->128->13 ----------------
__global__ void fp2fc_kernel(const half_t* __restrict__ x,
                             const half_t* __restrict__ wt1, const half_t* __restrict__ wt2,
                             const half_t* __restrict__ wt3, const half_t* __restrict__ wt4,
                             const half_t* __restrict__ wt5,
                             const float* __restrict__ b1, const float* __restrict__ b2,
                             const float* __restrict__ b3, const float* __restrict__ b4,
                             const float* __restrict__ b5,
                             float* __restrict__ out) {
  __shared__ half_t bufA[2][32 * 128];
  __shared__ half_t bufB[2][32 * 128];
  int wave = threadIdx.x >> 5;
  int lane = threadIdx.x & 31;
  int row0 = blockIdx.x * 64 + wave * 32;
  mlp_layer<2>(x + (size_t)row0 * 160, 160, wt1, 160, b1, 128, 128, bufA[wave], 128, true);
  __syncthreads();
  mlp_layer<2>(bufA[wave], 128, wt2, 128, b2, 128, 128, bufB[wave], 128, true);
  __syncthreads();
  mlp_layer<2>(bufB[wave], 128, wt3, 128, b3, 128, 128, bufA[wave], 128, true);
  __syncthreads();
  mlp_layer<2>(bufA[wave], 128, wt4, 128, b4, 128, 128, bufB[wave], 128, true);  // fc1
  __syncthreads();
  // fc2: 128 -> 16 (13 real), no relu, f32 out
  {
    v8f c[2];
    c[0] = (v8f){0.f,0.f,0.f,0.f,0.f,0.f,0.f,0.f};
    c[1] = (v8f){0.f,0.f,0.f,0.f,0.f,0.f,0.f,0.f};
    for (int kt = 0; kt < 128; kt += 32) {
      v16h bm = load_B(wt5, 128, 0, kt);
#pragma unroll
      for (int m = 0; m < 2; m++) {
        v16h am = load_A(bufB[wave], 128, m * 16, kt);
        c[m] = wmma16(am, bm, c[m]);
      }
    }
    int col = lane & 15;
    int rsel = (lane & 16) ? 8 : 0;
    if (col < 13) {
      float bb = b5[col];
#pragma unroll
      for (int m = 0; m < 2; m++)
#pragma unroll
        for (int i = 0; i < 8; i++)
          out[(size_t)(row0 + m * 16 + rsel + i) * 13 + col] = c[m][i] + bb;
    }
  }
}

// ---------------- host side ----------------
extern "C" void kernel_launch(void* const* d_in, const int* in_sizes, int n_in,
                              void* d_out, int out_size, void* d_ws, size_t ws_size,
                              hipStream_t stream) {
  (void)in_sizes; (void)n_in; (void)out_size; (void)ws_size;
  const float* xyz = (const float*)d_in[0];
  const float* sa1w[3] = {(const float*)d_in[1], (const float*)d_in[2], (const float*)d_in[3]};
  const float* sa1b[3] = {(const float*)d_in[4], (const float*)d_in[5], (const float*)d_in[6]};
  const float* sa2w[3] = {(const float*)d_in[7], (const float*)d_in[8], (const float*)d_in[9]};
  const float* sa2b[3] = {(const float*)d_in[10], (const float*)d_in[11], (const float*)d_in[12]};
  const float* fp1w[2] = {(const float*)d_in[13], (const float*)d_in[14]};
  const float* fp1b[2] = {(const float*)d_in[15], (const float*)d_in[16]};
  const float* fp2w[3] = {(const float*)d_in[17], (const float*)d_in[18], (const float*)d_in[19]};
  const float* fp2b[3] = {(const float*)d_in[20], (const float*)d_in[21], (const float*)d_in[22]};
  const float* fc1w = (const float*)d_in[23];
  const float* fc1b = (const float*)d_in[24];
  const float* fc2w = (const float*)d_in[25];
  const float* fc2b = (const float*)d_in[26];

  char* ws = (char*)d_ws;
  size_t off = 0;
  auto alloc = [&](size_t bytes) -> char* {
    char* p = ws + off;
    off = (off + bytes + 255) & ~(size_t)255;
    return p;
  };

  int*    fpsIdx1 = (int*)   alloc((size_t)BB * S1 * 4);
  float*  newXyz1 = (float*) alloc((size_t)BB * S1 * 3 * 4);
  int*    idx1    = (int*)   alloc((size_t)BB * S1 * K1 * 4);
  half_t* f1      = (half_t*)alloc((size_t)BB * S1 * 128 * 2);
  int*    fpsIdx2 = (int*)   alloc((size_t)BB * S2 * 4);
  float*  newXyz2 = (float*) alloc((size_t)BB * S2 * 3 * 4);
  int*    idx2    = (int*)   alloc((size_t)BB * S2 * K2 * 4);
  half_t* f2      = (half_t*)alloc((size_t)BB * S2 * 256 * 2);
  int*    knn1i   = (int*)   alloc((size_t)BB * S1 * 3 * 4);
  float*  knn1w   = (float*) alloc((size_t)BB * S1 * 3 * 4);
  half_t* xfp1    = (half_t*)alloc((size_t)BB * S1 * 384 * 2);
  half_t* up1     = (half_t*)alloc((size_t)BB * S1 * 128 * 2);
  int*    knn2i   = (int*)   alloc((size_t)BB * NN * 3 * 4);
  float*  knn2w   = (float*) alloc((size_t)BB * NN * 3 * 4);
  half_t* xfp2    = (half_t*)alloc((size_t)BB * NN * 160 * 2);

  // Transposed, padded f16 weights
  half_t* wtSA1[3]; half_t* wtSA2[3]; half_t* wtFP1[2]; half_t* wtFP2[3];
  half_t* wtFC1; half_t* wtFC2;
  wtSA1[0] = (half_t*)alloc((size_t)64 * 32 * 2);
  wtSA1[1] = (half_t*)alloc((size_t)64 * 64 * 2);
  wtSA1[2] = (half_t*)alloc((size_t)128 * 64 * 2);
  wtSA2[0] = (half_t*)alloc((size_t)128 * 160 * 2);
  wtSA2[1] = (half_t*)alloc((size_t)128 * 128 * 2);
  wtSA2[2] = (half_t*)alloc((size_t)256 * 128 * 2);
  wtFP1[0] = (half_t*)alloc((size_t)256 * 384 * 2);
  wtFP1[1] = (half_t*)alloc((size_t)128 * 256 * 2);
  wtFP2[0] = (half_t*)alloc((size_t)128 * 160 * 2);
  wtFP2[1] = (half_t*)alloc((size_t)128 * 128 * 2);
  wtFP2[2] = (half_t*)alloc((size_t)128 * 128 * 2);
  wtFC1    = (half_t*)alloc((size_t)128 * 128 * 2);
  wtFC2    = (half_t*)alloc((size_t)16 * 128 * 2);

  auto cdiv = [](int a, int b) { return (a + b - 1) / b; };
  auto prep = [&](const float* w, half_t* wt, int Cin, int Cout, int Cinp, int Coutp) {
    int total = Coutp * Cinp;
    prep_wt_kernel<<<cdiv(total, 256), 256, 0, stream>>>(w, wt, Cin, Cout, Cinp, Coutp);
  };
  prep(sa1w[0], wtSA1[0], 6, 64, 32, 64);
  prep(sa1w[1], wtSA1[1], 64, 64, 64, 64);
  prep(sa1w[2], wtSA1[2], 64, 128, 64, 128);
  prep(sa2w[0], wtSA2[0], 131, 128, 160, 128);
  prep(sa2w[1], wtSA2[1], 128, 128, 128, 128);
  prep(sa2w[2], wtSA2[2], 128, 256, 128, 256);
  prep(fp1w[0], wtFP1[0], 384, 256, 384, 256);
  prep(fp1w[1], wtFP1[1], 256, 128, 256, 128);
  prep(fp2w[0], wtFP2[0], 131, 128, 160, 128);
  prep(fp2w[1], wtFP2[1], 128, 128, 128, 128);
  prep(fp2w[2], wtFP2[2], 128, 128, 128, 128);
  prep(fc1w, wtFC1, 128, 128, 128, 128);
  prep(fc2w, wtFC2, 128, 13, 128, 16);

  // ---- SA1 ----
  fps_kernel<<<BB, 256, 0, stream>>>(xyz, NN, S1, fpsIdx1, newXyz1);
  ball_query_kernel<<<BB * S1, 32, 0, stream>>>(xyz, NN, newXyz1, S1, 0.2f * 0.2f, K1, idx1);
  sa1_kernel<<<BB * S1 / 4, 128, 0, stream>>>(xyz, idx1, newXyz1,
                                              wtSA1[0], wtSA1[1], wtSA1[2],
                                              sa1b[0], sa1b[1], sa1b[2], f1);
  // ---- SA2 ----
  fps_kernel<<<BB, 256, 0, stream>>>(newXyz1, S1, S2, fpsIdx2, newXyz2);
  ball_query_kernel<<<BB * S2, 32, 0, stream>>>(newXyz1, S1, newXyz2, S2, 0.4f * 0.4f, K2, idx2);
  sa2_kernel<<<BB * S2, 32, 0, stream>>>(newXyz1, f1, idx2, newXyz2,
                                         wtSA2[0], wtSA2[1], wtSA2[2],
                                         sa2b[0], sa2b[1], sa2b[2], f2);
  // ---- FP1: xyz1 <- xyz2 ----
  knn3_kernel<<<cdiv(BB * S1, 256), 256, 0, stream>>>(newXyz1, S1, newXyz2, S2, BB * S1, knn1i, knn1w);
  build_fp1_x_kernel<<<cdiv(BB * S1 * 384, 256), 256, 0, stream>>>(f1, f2, knn1i, knn1w, xfp1);
  fp1_kernel<<<BB * S1 / 64, 64, 0, stream>>>(xfp1, wtFP1[0], wtFP1[1], fp1b[0], fp1b[1], up1);
  // ---- FP2 + FC: xyz <- xyz1 ----
  knn3_kernel<<<cdiv(BB * NN, 256), 256, 0, stream>>>(xyz, NN, newXyz1, S1, BB * NN, knn2i, knn2w);
  build_fp2_x_kernel<<<cdiv(BB * NN * 160, 256), 256, 0, stream>>>(xyz, up1, knn2i, knn2w, xfp2);
  fp2fc_kernel<<<BB * NN / 64, 64, 0, stream>>>(xfp2, wtFP2[0], wtFP2[1], wtFP2[2], wtFC1, wtFC2,
                                                fp2b[0], fp2b[1], fp2b[2], fc1b, fc2b,
                                                (float*)d_out);
}
